// CompressedLinearRANS_4002909520069
// MI455X (gfx1250) — matrix operand order repaired and checked by reading, not compiled, not run on hardware
//
#include <hip/hip_runtime.h>

// ---------- types ----------
typedef __attribute__((ext_vector_type(16))) __bf16       bf16x16;
typedef __attribute__((ext_vector_type(8)))  __bf16       bf16x8;
typedef __attribute__((ext_vector_type(8)))  float        f32x8;
typedef __attribute__((ext_vector_type(8)))  int          i32x8;
typedef __attribute__((ext_vector_type(4)))  float        f32x4;
typedef __attribute__((ext_vector_type(4)))  int          i32x4;
typedef __attribute__((ext_vector_type(4)))  unsigned int u32x4;
typedef __attribute__((ext_vector_type(2)))  unsigned int u32x2;

// ---------- problem size (B=4,S=2048 -> M=8192; OUT=4096; IN=4096) ----------
#define M_TOT 8192
#define N_TOT 4096
#define K_TOT 4096

// ---------- CDNA5 async global->LDS availability (probe-confirmed present) ----------
#if defined(__has_builtin)
#  if __has_builtin(__builtin_amdgcn_global_load_async_to_lds_b128) && \
      __has_builtin(__builtin_amdgcn_s_wait_asynccnt)
#    define USE_ASYNC_LDS 1
#  endif
#endif
#ifndef USE_ASYNC_LDS
#  define USE_ASYNC_LDS 0
#endif

// builtin expects v4i pointers: (global v4i* src, local v4i* dst, imm offset, imm cpol)
#define GPTR(p) ((__attribute__((address_space(1))) i32x4*)(p))
#define LPTR(p) ((__attribute__((address_space(3))) i32x4*)(p))

// =====================================================================
// Pass 1: one-time dtype conversion into workspace (bf16, same layouts)
// =====================================================================
__global__ __launch_bounds__(256)
void cvt_x_bf16(const float* __restrict__ in, unsigned short* __restrict__ out) {
    const size_t i = ((size_t)blockIdx.x * 256 + threadIdx.x) * 8;
    f32x8 v = *(const f32x8*)(in + i);
    union { bf16x8 h; u32x4 q; } r;
    r.h = __builtin_convertvector(v, bf16x8);     // RNE
    *(u32x4*)(out + i) = r.q;
}

__global__ __launch_bounds__(256)
void cvt_w_bf16(const int* __restrict__ in, unsigned short* __restrict__ out) {
    const size_t i = ((size_t)blockIdx.x * 256 + threadIdx.x) * 8;
    i32x8 v = *(const i32x8*)(in + i);
    union { bf16x8 h; u32x4 q; } r;
    r.h = __builtin_convertvector(v, bf16x8);     // |v|<=127 -> exact in bf16
    *(u32x4*)(out + i) = r.q;
}

// =====================================================================
// Pass 2: pure-bf16 WMMA GEMM
//   block tile 128x128, k-stage 64, 8 waves (4M x 2N), wave tile 32x64
//   -> 16x v_wmma_f32_16x16x32_bf16 per stage
//   feed path: async global->LDS (double-buffered) when available
// =====================================================================
#define TILE_M 128
#define TILE_N 128
#define TILE_K 64
#define KSTEPS (K_TOT / TILE_K)
#define LDSS   72            // LDS row stride in bf16 (64 + 8 pad -> conflict-free b128)

__global__ __launch_bounds__(256)
void gemm_wmma_bf16(const unsigned short* __restrict__ Xb,  // [M_TOT,K_TOT] bf16
                    const unsigned short* __restrict__ Wb,  // [N_TOT,K_TOT] bf16
                    const float* __restrict__ scale_p,
                    const float* __restrict__ bias,
                    float*       __restrict__ Out)
{
#if USE_ASYNC_LDS
    __shared__ __align__(16) unsigned short As[2][TILE_M * LDSS];
    __shared__ __align__(16) unsigned short Bs[2][TILE_N * LDSS];
#else
    __shared__ __align__(16) unsigned short As[1][TILE_M * LDSS];
    __shared__ __align__(16) unsigned short Bs[1][TILE_N * LDSS];
#endif

    const int t      = threadIdx.x;
    const int lane   = t & 31;
    const int wv     = t >> 5;
    const int lane16 = lane & 15;
    const int half   = lane >> 4;
    const int waveM  = (wv & 3) * 32;   // 4 waves along M
    const int waveN  = (wv >> 2) * 64;  // 2 waves along N
    const int mBase  = blockIdx.y * TILE_M;
    const int nBase  = blockIdx.x * TILE_N;

    // cooperative load mapping: 256 threads x 4 x b128(8 bf16) = 128x64 tile
    const int ldRow = t >> 3;           // 0..31 (+ i*32)
    const int ldCol = (t & 7) * 8;      // 0,8,...,56

    const unsigned short* xp = Xb + (size_t)(mBase + ldRow) * K_TOT + ldCol;
    const unsigned short* wp = Wb + (size_t)(nBase + ldRow) * K_TOT + ldCol;

    f32x8 acc[2][4] = {};

#if USE_ASYNC_LDS
    // ---- async double-buffered pipeline ----
    auto issue_stage = [&](int ks, int buf) {
        const unsigned short* xg = xp + (size_t)ks * TILE_K;
        const unsigned short* wg = wp + (size_t)ks * TILE_K;
        unsigned short* al = &As[buf][ldRow * LDSS + ldCol];
        unsigned short* bl = &Bs[buf][ldRow * LDSS + ldCol];
#pragma unroll
        for (int i = 0; i < 4; ++i) {
            __builtin_amdgcn_global_load_async_to_lds_b128(
                GPTR(xg + (size_t)i * 32 * K_TOT), LPTR(al + i * 32 * LDSS), 0, 0);
            __builtin_amdgcn_global_load_async_to_lds_b128(
                GPTR(wg + (size_t)i * 32 * K_TOT), LPTR(bl + i * 32 * LDSS), 0, 0);
        }
    };

    issue_stage(0, 0);
    issue_stage(1, 1);

    for (int kt = 0; kt < KSTEPS; ++kt) {
        const int buf = kt & 1;

        // stage kt's 8 async loads are the oldest outstanding (in-order done)
        if (kt + 1 < KSTEPS) __builtin_amdgcn_s_wait_asynccnt(8);
        else                 __builtin_amdgcn_s_wait_asynccnt(0);
        __syncthreads();                       // all waves' tile data visible

#pragma unroll
        for (int ks = 0; ks < 2; ++ks) {
            union Frag { bf16x16 v; u32x4 q[2]; } a[2], b[4];
#pragma unroll
            for (int mi = 0; mi < 2; ++mi) {
                const int r = waveM + mi * 16 + lane16;
                const unsigned short* p = &As[buf][r * LDSS + ks * 32 + half * 8];
                a[mi].q[0] = *(const u32x4*)p;         // K {0..7}   | {8..15}
                a[mi].q[1] = *(const u32x4*)(p + 16);  // K {16..23} | {24..31}
            }
#pragma unroll
            for (int ni = 0; ni < 4; ++ni) {
                const int r = waveN + ni * 16 + lane16;
                const unsigned short* p = &Bs[buf][r * LDSS + ks * 32 + half * 8];
                b[ni].q[0] = *(const u32x4*)p;         // K {0..7}   | {8..15}
                b[ni].q[1] = *(const u32x4*)(p + 16);  // K {16..23} | {24..31}
            }
#pragma unroll
            for (int mi = 0; mi < 2; ++mi)
#pragma unroll
                for (int ni = 0; ni < 4; ++ni)
                    acc[mi][ni] = __builtin_amdgcn_wmma_f32_16x16x32_bf16(
                        false, a[mi].v, false, b[ni].v,
                        (short)0, acc[mi][ni], false, false);
        }

        __syncthreads();                       // everyone done reading LDS[buf]
        if (kt + 2 < KSTEPS) issue_stage(kt + 2, buf);  // overlaps next stage's WMMAs
    }
#else
    // ---- fallback: register-staged prefetch (known-good) ----
    u32x4 ra[4], rb[4];
#pragma unroll
    for (int i = 0; i < 4; ++i) {
        ra[i] = *(const u32x4*)(xp + (size_t)i * 32 * K_TOT);
        rb[i] = *(const u32x4*)(wp + (size_t)i * 32 * K_TOT);
    }
    for (int kt = 0; kt < KSTEPS; ++kt) {
#pragma unroll
        for (int i = 0; i < 4; ++i) {
            const int r = ldRow + i * 32;
            *(u32x4*)&As[0][r * LDSS + ldCol] = ra[i];
            *(u32x4*)&Bs[0][r * LDSS + ldCol] = rb[i];
        }
        __syncthreads();
        if (kt + 1 < KSTEPS) {
            const unsigned short* xn = xp + (size_t)(kt + 1) * TILE_K;
            const unsigned short* wn = wp + (size_t)(kt + 1) * TILE_K;
#pragma unroll
            for (int i = 0; i < 4; ++i) {
                ra[i] = *(const u32x4*)(xn + (size_t)i * 32 * K_TOT);
                rb[i] = *(const u32x4*)(wn + (size_t)i * 32 * K_TOT);
            }
        }
#pragma unroll
        for (int ks = 0; ks < 2; ++ks) {
            union Frag { bf16x16 v; u32x4 q[2]; } a[2], b[4];
#pragma unroll
            for (int mi = 0; mi < 2; ++mi) {
                const int r = waveM + mi * 16 + lane16;
                const unsigned short* p = &As[0][r * LDSS + ks * 32 + half * 8];
                a[mi].q[0] = *(const u32x4*)p;
                a[mi].q[1] = *(const u32x4*)(p + 16);
            }
#pragma unroll
            for (int ni = 0; ni < 4; ++ni) {
                const int r = waveN + ni * 16 + lane16;
                const unsigned short* p = &Bs[0][r * LDSS + ks * 32 + half * 8];
                b[ni].q[0] = *(const u32x4*)p;
                b[ni].q[1] = *(const u32x4*)(p + 16);
            }
#pragma unroll
            for (int mi = 0; mi < 2; ++mi)
#pragma unroll
                for (int ni = 0; ni < 4; ++ni)
                    acc[mi][ni] = __builtin_amdgcn_wmma_f32_16x16x32_bf16(
                        false, a[mi].v, false, b[ni].v,
                        (short)0, acc[mi][ni], false, false);
        }
        __syncthreads();
    }
#endif

    // ---- epilogue: dequant scale + bias ----
    const float sc = scale_p[0];
    float bv[4];
#pragma unroll
    for (int ni = 0; ni < 4; ++ni)
        bv[ni] = bias[nBase + waveN + ni * 16 + lane16];

#pragma unroll
    for (int mi = 0; mi < 2; ++mi) {
#pragma unroll
        for (int ni = 0; ni < 4; ++ni) {
            const int col = nBase + waveN + ni * 16 + lane16;
#pragma unroll
            for (int r = 0; r < 8; ++r) {
                const int row = mBase + waveM + mi * 16 + half * 8 + r;
                __builtin_nontemporal_store(acc[mi][ni][r] * sc + bv[ni],
                                            Out + (size_t)row * N_TOT + col);
            }
        }
    }
}

// =====================================================================
// Fallback (ws too small): fused single-pass kernel
// =====================================================================
#define FTILE_K 32
#define FLDSS   40

static __device__ __forceinline__ unsigned int bf16pk(float a, float b) {
    unsigned int ua = __builtin_bit_cast(unsigned int, a);
    unsigned int ub = __builtin_bit_cast(unsigned int, b);
    ua = (ua + 0x7FFFu + ((ua >> 16) & 1u)) >> 16;
    ub = (ub + 0x7FFFu + ((ub >> 16) & 1u)) & 0xFFFF0000u;
    return ua | ub;
}

__global__ __launch_bounds__(256)
void fused_gemm_wmma(const float* __restrict__ X, const int* __restrict__ W,
                     const float* __restrict__ scale_p, const float* __restrict__ bias,
                     float* __restrict__ Out)
{
    __shared__ __align__(16) unsigned short As[TILE_M * FLDSS];
    __shared__ __align__(16) unsigned short Bs[TILE_N * FLDSS];

    const int t      = threadIdx.x;
    const int lane   = t & 31;
    const int wv     = t >> 5;
    const int lane16 = lane & 15;
    const int half   = lane >> 4;
    const int waveM  = (wv & 3) * 32;
    const int waveN  = (wv >> 2) * 64;
    const int mBase  = blockIdx.y * TILE_M;
    const int nBase  = blockIdx.x * TILE_N;

    const int ldRow = t >> 3;
    const int ldCol = (t & 7) * 4;

    const float* xp = X + (size_t)(mBase + ldRow) * K_TOT + ldCol;
    const int*   wp = W + (size_t)(nBase + ldRow) * K_TOT + ldCol;

    f32x4 ra[4];
    i32x4 rb[4];
#pragma unroll
    for (int i = 0; i < 4; ++i) {
        ra[i] = *(const f32x4*)(xp + (size_t)i * 32 * K_TOT);
        rb[i] = *(const i32x4*)(wp + (size_t)i * 32 * K_TOT);
    }

    f32x8 acc[2][4] = {};

    for (int kt = 0; kt < K_TOT / FTILE_K; ++kt) {
#pragma unroll
        for (int i = 0; i < 4; ++i) {
            const int r = ldRow + i * 32;
            u32x2 pa; pa[0] = bf16pk(ra[i][0], ra[i][1]); pa[1] = bf16pk(ra[i][2], ra[i][3]);
            *(u32x2*)&As[r * FLDSS + ldCol] = pa;
            u32x2 pb; pb[0] = bf16pk((float)rb[i][0], (float)rb[i][1]);
            pb[1] = bf16pk((float)rb[i][2], (float)rb[i][3]);
            *(u32x2*)&Bs[r * FLDSS + ldCol] = pb;
        }
        __syncthreads();

        if (kt + 1 < K_TOT / FTILE_K) {
            const float* xn = xp + (size_t)(kt + 1) * FTILE_K;
            const int*   wn = wp + (size_t)(kt + 1) * FTILE_K;
#pragma unroll
            for (int i = 0; i < 4; ++i) {
                ra[i] = *(const f32x4*)(xn + (size_t)i * 32 * K_TOT);
                rb[i] = *(const i32x4*)(wn + (size_t)i * 32 * K_TOT);
            }
        }

        union Frag { bf16x16 v; u32x4 q[2]; } a[2], b[4];
#pragma unroll
        for (int mi = 0; mi < 2; ++mi) {
            const int r = waveM + mi * 16 + lane16;
            const unsigned short* p = &As[r * FLDSS + half * 8];
            a[mi].q[0] = *(const u32x4*)p;
            a[mi].q[1] = *(const u32x4*)(p + 16);
        }
#pragma unroll
        for (int ni = 0; ni < 4; ++ni) {
            const int r = waveN + ni * 16 + lane16;
            const unsigned short* p = &Bs[r * FLDSS + half * 8];
            b[ni].q[0] = *(const u32x4*)p;
            b[ni].q[1] = *(const u32x4*)(p + 16);
        }
#pragma unroll
        for (int mi = 0; mi < 2; ++mi)
#pragma unroll
            for (int ni = 0; ni < 4; ++ni)
                acc[mi][ni] = __builtin_amdgcn_wmma_f32_16x16x32_bf16(
                    false, a[mi].v, false, b[ni].v, (short)0, acc[mi][ni], false, false);
        __syncthreads();
    }

    const float sc = scale_p[0];
    float bv[4];
#pragma unroll
    for (int ni = 0; ni < 4; ++ni)
        bv[ni] = bias[nBase + waveN + ni * 16 + lane16];
#pragma unroll
    for (int mi = 0; mi < 2; ++mi)
#pragma unroll
        for (int ni = 0; ni < 4; ++ni) {
            const int col = nBase + waveN + ni * 16 + lane16;
#pragma unroll
            for (int r = 0; r < 8; ++r) {
                const int row = mBase + waveM + mi * 16 + half * 8 + r;
                __builtin_nontemporal_store(acc[mi][ni][r] * sc + bv[ni],
                                            Out + (size_t)row * N_TOT + col);
            }
        }
}

// =====================================================================
extern "C" void kernel_launch(void* const* d_in, const int* in_sizes, int n_in,
                              void* d_out, int out_size, void* d_ws, size_t ws_size,
                              hipStream_t stream) {
    (void)in_sizes; (void)n_in; (void)out_size;
    const float* x     = (const float*)d_in[0];
    const int*   w     = (const int*)  d_in[1];
    const float* scale = (const float*)d_in[2];
    const float* bias  = (const float*)d_in[3];
    float*       out   = (float*)d_out;

    const size_t xElems = (size_t)M_TOT * K_TOT;
    const size_t wElems = (size_t)N_TOT * K_TOT;
    const size_t needWs = (xElems + wElems) * sizeof(unsigned short);  // 96 MiB

    dim3 grid(N_TOT / TILE_N, M_TOT / TILE_M);            // (32, 64)

    if (ws_size >= needWs) {
        unsigned short* xb = (unsigned short*)d_ws;
        unsigned short* wb = xb + xElems;
        cvt_x_bf16<<<(unsigned)(xElems / (256 * 8)), 256, 0, stream>>>(x, xb);
        cvt_w_bf16<<<(unsigned)(wElems / (256 * 8)), 256, 0, stream>>>(w, wb);
        gemm_wmma_bf16<<<grid, 256, 0, stream>>>(xb, wb, scale, bias, out);
    } else {
        fused_gemm_wmma<<<grid, 256, 0, stream>>>(x, w, scale, bias, out);
    }
}
